// AttentionBlock_68075231641721
// MI455X (gfx1250) — matrix operand order repaired and checked by reading, not compiled, not run on hardware
//
#include <hip/hip_runtime.h>
#include <hip/hip_bf16.h>

// ---------------------------------------------------------------------------
// Types for CDNA5 WMMA bf16 (16x16x32, f32 accumulate)
// ---------------------------------------------------------------------------
typedef __attribute__((ext_vector_type(16))) __bf16 bf16x16;
typedef __attribute__((ext_vector_type(8)))  __bf16 bf16x8;
typedef __attribute__((ext_vector_type(8)))  float  f32x8;

#define WMMA_BF16(A, B, C) \
  __builtin_amdgcn_wmma_f32_16x16x32_bf16(false, (A), false, (B), (short)0, (C), false, false)

// ---------------------------------------------------------------------------
// CDNA5 async global->LDS copy (ASYNCcnt-tracked DMA; ISA ch.10 / 15.18.3).
// VDST VGPR carries the LDS byte offset (generic LDS address truncates to the
// LDS offset in addr[31:0], per the aperture mapping table).
// ---------------------------------------------------------------------------
__device__ __forceinline__ unsigned lds_off32(const void* p) {
  return (unsigned)(uintptr_t)p;
}
__device__ __forceinline__ void async_load_b128(unsigned lds_off, const void* gptr) {
  asm volatile("global_load_async_to_lds_b128 %0, %1, off"
               :: "v"(lds_off), "v"((unsigned long long)(uintptr_t)gptr)
               : "memory");
}
__device__ __forceinline__ void wait_async0() {
  asm volatile("s_wait_asynccnt 0x0" ::: "memory");
}

// ---------------------------------------------------------------------------
// Fragment loaders. Tiles are row-major with K contiguous, row stride = ld.
// A (16x32, M x K): lane L holds row M = L%16; 16 contiguous K values
//   [kb..kb+15] where kb = (L/16)*16 (matches B fragment K order).
// B (32x16, K x N): lane L holds column N = L%16 (tile stores B^T rows);
//   16 contiguous K values starting at (L/16)*16 (same K order as the A fragment).
// ---------------------------------------------------------------------------
__device__ __forceinline__ bf16x16 load_frag_a(const __bf16* tile, int ld) {
  const int lane = threadIdx.x & 31;
  const int row  = lane & 15;
  const int kb   = (lane >> 4) * 16;
  const __bf16* p = tile + row * ld + kb;
  bf16x8 lo = *(const bf16x8*)(p);
  bf16x8 hi = *(const bf16x8*)(p + 8);
  bf16x16 r;
#pragma unroll
  for (int i = 0; i < 8; ++i) { r[i] = lo[i]; r[i + 8] = hi[i]; }
  return r;
}

__device__ __forceinline__ bf16x16 load_frag_b(const __bf16* tile, int ld) {
  const int lane = threadIdx.x & 31;
  const int row  = lane & 15;
  const int kb   = (lane >> 4) * 16;
  return *(const bf16x16*)(tile + row * ld + kb);
}

// ---------------------------------------------------------------------------
// Problem constants
// ---------------------------------------------------------------------------
#define BB 4
#define LL 1024
#define CC 768
#define HH 12
#define DD 64
#define ML (BB * LL)        // 4096 rows for projection GEMMs

// ---------------------------------------------------------------------------
// K0a: flat f32 -> bf16 convert
// ---------------------------------------------------------------------------
__global__ void k_cvt(const float* __restrict__ in, __bf16* __restrict__ out, int n) {
  int i = blockIdx.x * blockDim.x + threadIdx.x;
  if (i < n) out[i] = (__bf16)in[i];
}

// K0b: transpose + convert: out[n*K + k] = in[k*N + n]
__global__ void k_transpose_cvt(const float* __restrict__ in, __bf16* __restrict__ out,
                                int K, int N) {
  int i = blockIdx.x * blockDim.x + threadIdx.x;
  if (i >= K * N) return;
  int n = i / K, k = i - n * K;
  out[i] = (__bf16)in[k * N + n];
}

// ---------------------------------------------------------------------------
// K1/K5: tiled bf16 WMMA GEMM with double-buffered ASYNC global->LDS staging.
// C[M x 768] = A[M x Kd] * Bt[768 x Kd]^T
// BM = BN = 64, BK = 32; 128 threads (4 waves); wave w owns rows 16w..16w+15.
// mode 0: Q epilogue  (scale 1/8, store [B,H,L,D] bf16)
// mode 1: K epilogue  (store [B,H,L,D] bf16)
// mode 2: V epilogue  (store transposed [B,H,D,L] bf16)
// mode 3: f32 row-major store (final output projection)
// ---------------------------------------------------------------------------
__global__ __launch_bounds__(128)
void k_gemm_bf16(const __bf16* __restrict__ A, const __bf16* __restrict__ Bt,
                 int Kd, __bf16* __restrict__ outB, float* __restrict__ outF, int mode) {
  __shared__ __align__(64) __bf16 sA[2][64 * 32];
  __shared__ __align__(64) __bf16 sB[2][64 * 32];
  const int tid = threadIdx.x, wave = tid >> 5, lane = tid & 31;
  const int m0 = blockIdx.y * 64;
  const int n0 = blockIdx.x * 64;

  f32x8 acc[4] = {};

  auto stage = [&](int buf, int k0) {
#pragma unroll
    for (int s = 0; s < 2; ++s) {
      int idx = tid + s * 128;            // 256 16B chunks per tile
      int r = idx >> 2;
      int c8 = (idx & 3) * 8;
      async_load_b128(lds_off32(&sA[buf][r * 32 + c8]),
                      &A[(size_t)(m0 + r) * Kd + k0 + c8]);
      async_load_b128(lds_off32(&sB[buf][r * 32 + c8]),
                      &Bt[(size_t)(n0 + r) * Kd + k0 + c8]);
    }
  };

  const int nT = Kd / 32;
  stage(0, 0);
  wait_async0();
  __syncthreads();

  for (int t = 0; t < nT; ++t) {
    const int cur = t & 1;
    if (t + 1 < nT) stage(1 - cur, (t + 1) * 32);   // overlap DMA with WMMA

    bf16x16 af = load_frag_a(&sA[cur][(wave * 16) * 32], 32);
#pragma unroll
    for (int j = 0; j < 4; ++j) {
      bf16x16 bfrag = load_frag_b(&sB[cur][(j * 16) * 32], 32);
      acc[j] = WMMA_BF16(af, bfrag, acc[j]);
    }

    if (t + 1 < nT) wait_async0();
    __syncthreads();
  }

  const float scale = (mode == 0) ? 0.125f : 1.0f;  // 1/sqrt(64)
#pragma unroll
  for (int j = 0; j < 4; ++j) {
#pragma unroll
    for (int jj = 0; jj < 8; ++jj) {
      int row = m0 + wave * 16 + jj + ((lane >> 4) << 3);   // C layout: M
      int col = n0 + j * 16 + (lane & 15);                  // C layout: N
      float v = acc[j][jj] * scale;
      int b = row >> 10, l = row & 1023, h = col >> 6, d = col & 63;
      if (mode <= 1) {
        outB[(((size_t)b * HH + h) * LL + l) * DD + d] = (__bf16)v;
      } else if (mode == 2) {
        outB[(((size_t)b * HH + h) * DD + d) * LL + l] = (__bf16)v;
      } else {
        outF[(size_t)row * CC + col] = v;
      }
    }
  }
}

// ---------------------------------------------------------------------------
// K2: attention logits for all 12 heads + pre-softmax talking-heads mix.
// Grid: (L/64 key-tiles = 16, L/16 query-tiles = 64, B). Block 128 (4 waves).
// Wave w covers key sub-tile 16w within the 64-key tile.
// Q/K tiles staged via async DMA.
// ---------------------------------------------------------------------------
__global__ __launch_bounds__(128)
void k_logits_premix(const __bf16* __restrict__ Qh, const __bf16* __restrict__ Kh,
                     const float* __restrict__ Wpre, __bf16* __restrict__ Smix) {
  __shared__ __align__(64) __bf16 sQ[16 * 64];
  __shared__ __align__(64) __bf16 sK[64 * 64];
  const int tid = threadIdx.x, wave = tid >> 5, lane = tid & 31;
  const int kt = blockIdx.x, lt = blockIdx.y, b = blockIdx.z;

  f32x8 acc[HH] = {};
#pragma unroll
  for (int h = 0; h < HH; ++h) {
    const __bf16* qsrc = Qh + (((size_t)b * HH + h) * LL + lt * 16) * DD;
    const __bf16* ksrc = Kh + (((size_t)b * HH + h) * LL + kt * 64) * DD;
    {
      int r = tid >> 3, c8 = (tid & 7) * 8;                 // 128 chunks -> 16x64
      async_load_b128(lds_off32(&sQ[r * 64 + c8]), &qsrc[(size_t)r * DD + c8]);
    }
#pragma unroll
    for (int s = 0; s < 4; ++s) {
      int idx = tid + s * 128;                              // 512 chunks -> 64x64
      int r = idx >> 3, c8 = (idx & 7) * 8;
      async_load_b128(lds_off32(&sK[r * 64 + c8]), &ksrc[(size_t)r * DD + c8]);
    }
    wait_async0();
    __syncthreads();
#pragma unroll
    for (int kk = 0; kk < 64; kk += 32) {
      bf16x16 af    = load_frag_a(&sQ[kk], 64);
      bf16x16 bfrag = load_frag_b(&sK[(wave * 16) * 64 + kk], 64);
      acc[h] = WMMA_BF16(af, bfrag, acc[h]);
    }
    __syncthreads();
  }

  // pre-softmax mix: m_i = sum_h s_h * W_pre[h,i]
#pragma unroll
  for (int i = 0; i < HH; ++i) {
    f32x8 m = {};
#pragma unroll
    for (int h = 0; h < HH; ++h) {
      float w = Wpre[h * HH + i];
#pragma unroll
      for (int jj = 0; jj < 8; ++jj) m[jj] += acc[h][jj] * w;
    }
#pragma unroll
    for (int jj = 0; jj < 8; ++jj) {
      int l = lt * 16 + jj + ((lane >> 4) << 3);
      int k = kt * 64 + wave * 16 + (lane & 15);
      Smix[(((size_t)b * HH + i) * LL + l) * LL + k] = (__bf16)m[jj];
    }
  }
}

// ---------------------------------------------------------------------------
// K3: row softmax over k (length 1024), in place, one wave32 per row.
// ---------------------------------------------------------------------------
__global__ __launch_bounds__(256)
void k_softmax(__bf16* __restrict__ P) {
  const int wave = threadIdx.x >> 5, lane = threadIdx.x & 31;
  const size_t row = (size_t)blockIdx.x * 8 + wave;
  __bf16* p = P + row * LL;

  float v[32];
  float mx = -3.0e38f;
#pragma unroll
  for (int j = 0; j < 32; ++j) {
    v[j] = (float)p[lane + 32 * j];
    mx = fmaxf(mx, v[j]);
  }
#pragma unroll
  for (int off = 16; off >= 1; off >>= 1) mx = fmaxf(mx, __shfl_xor(mx, off, 32));
  float sum = 0.f;
#pragma unroll
  for (int j = 0; j < 32; ++j) { v[j] = __expf(v[j] - mx); sum += v[j]; }
#pragma unroll
  for (int off = 16; off >= 1; off >>= 1) sum += __shfl_xor(sum, off, 32);
  float inv = 1.0f / sum;
#pragma unroll
  for (int j = 0; j < 32; ++j) p[lane + 32 * j] = (__bf16)(v[j] * inv);
}

// ---------------------------------------------------------------------------
// K4: post-softmax talking-heads mix + PV. Grid (L/16 = 64, B); block 128.
// Wave w owns output-channel sub-tile d = 16w..16w+15 (D = 64).
// P tiles staged via async DMA; V^T fragments streamed straight from global.
// ---------------------------------------------------------------------------
__global__ __launch_bounds__(128)
void k_postmix_pv(const __bf16* __restrict__ P, const __bf16* __restrict__ Vt,
                  const float* __restrict__ Wpost, __bf16* __restrict__ Sout) {
  __shared__ __align__(64) __bf16 sP[HH * 16 * 64];   // 24 KB
  __shared__ __align__(64) __bf16 sW[HH * 16 * 64];   // 24 KB
  const int tid = threadIdx.x, wave = tid >> 5, lane = tid & 31;
  const int lt = blockIdx.x, b = blockIdx.y;

  f32x8 acc[HH] = {};
  for (int kt = 0; kt < 16; ++kt) {
#pragma unroll
    for (int h = 0; h < HH; ++h) {
      int r = tid >> 3, c8 = (tid & 7) * 8;             // 16x64 tile per head
      const __bf16* src = P + (((size_t)b * HH + h) * LL + lt * 16) * LL + kt * 64;
      async_load_b128(lds_off32(&sP[h * 1024 + r * 64 + c8]),
                      &src[(size_t)r * LL + c8]);
    }
    wait_async0();
    __syncthreads();
    // post-mix in LDS: wm_i = sum_h p_h * W_post[h,i]
#pragma unroll
    for (int e = 0; e < 8; ++e) {
      int el = tid * 8 + e;                             // 0..1023 within [16][64]
      float ph[HH];
#pragma unroll
      for (int h = 0; h < HH; ++h) ph[h] = (float)sP[h * 1024 + el];
#pragma unroll
      for (int i = 0; i < HH; ++i) {
        float s = 0.f;
#pragma unroll
        for (int h = 0; h < HH; ++h) s += ph[h] * Wpost[h * HH + i];
        sW[i * 1024 + el] = (__bf16)s;
      }
    }
    __syncthreads();
#pragma unroll
    for (int i = 0; i < HH; ++i) {
#pragma unroll
      for (int kk = 0; kk < 64; kk += 32) {
        bf16x16 af = load_frag_a(&sW[i * 1024 + kk], 64);
        int d = wave * 16 + (lane & 15);
        int kb = (lane >> 4) * 16;
        const __bf16* vp = Vt + (((size_t)b * HH + i) * DD + d) * LL + kt * 64 + kk + kb;
        bf16x16 bfrag = *(const bf16x16*)vp;            // B frag straight from global
        acc[i] = WMMA_BF16(af, bfrag, acc[i]);
      }
    }
    if (kt + 1 < 16) {                                  // prefetch next V^T tiles
      int d = wave * 16 + (lane & 15);
      __builtin_prefetch(&Vt[(((size_t)b * HH) * DD + d) * LL + (kt + 1) * 64], 0, 3);
    }
    __syncthreads();
  }

#pragma unroll
  for (int i = 0; i < HH; ++i) {
#pragma unroll
    for (int jj = 0; jj < 8; ++jj) {
      int l = lt * 16 + jj + ((lane >> 4) << 3);
      int d = wave * 16 + (lane & 15);
      Sout[((size_t)b * LL + l) * CC + i * DD + d] = (__bf16)acc[i][jj];
    }
  }
}

// ---------------------------------------------------------------------------
// Host launch
// ---------------------------------------------------------------------------
extern "C" void kernel_launch(void* const* d_in, const int* in_sizes, int n_in,
                              void* d_out, int out_size, void* d_ws, size_t ws_size,
                              hipStream_t stream) {
  const float* inq   = (const float*)d_in[0];
  const float* inkv  = (const float*)d_in[1];
  const float* Wq    = (const float*)d_in[2];
  const float* Wk    = (const float*)d_in[3];
  const float* Wv    = (const float*)d_in[4];
  const float* Wout  = (const float*)d_in[5];
  const float* Wpre  = (const float*)d_in[6];
  const float* Wpost = (const float*)d_in[7];

  char* ws = (char*)d_ws;
  size_t off = 0;
  auto carve = [&](size_t bytes) -> __bf16* {
    __bf16* p = (__bf16*)(ws + off);
    off += (bytes + 255) & ~(size_t)255;
    return p;
  };

  const size_t nX  = (size_t)ML * CC;          // 4096*768
  const size_t nW  = (size_t)CC * CC;          // 768*768
  const size_t nH  = (size_t)BB * HH * LL * DD;
  const size_t nS  = (size_t)BB * HH * LL * LL;

  __bf16* XqB  = carve(nX * 2);
  __bf16* XkvB = carve(nX * 2);
  __bf16* WqT  = carve(nW * 2);
  __bf16* WkT  = carve(nW * 2);
  __bf16* WvT  = carve(nW * 2);
  __bf16* WoT  = carve(nW * 2);
  __bf16* Qh   = carve(nH * 2);
  __bf16* Kh   = carve(nH * 2);
  __bf16* Vt   = carve(nH * 2);
  __bf16* Smix = carve(nS * 2);
  __bf16* SoB  = carve(nX * 2);

  // Stage 0: conversions
  k_cvt<<<(int)((nX + 255) / 256), 256, 0, stream>>>(inq, XqB, (int)nX);
  k_cvt<<<(int)((nX + 255) / 256), 256, 0, stream>>>(inkv, XkvB, (int)nX);
  k_transpose_cvt<<<(int)((nW + 255) / 256), 256, 0, stream>>>(Wq,   WqT, CC, CC);
  k_transpose_cvt<<<(int)((nW + 255) / 256), 256, 0, stream>>>(Wk,   WkT, CC, CC);
  k_transpose_cvt<<<(int)((nW + 255) / 256), 256, 0, stream>>>(Wv,   WvT, CC, CC);
  k_transpose_cvt<<<(int)((nW + 255) / 256), 256, 0, stream>>>(Wout, WoT, CC, CC);

  // Stage 1: QKV projections (WMMA + async DMA staging)
  dim3 ggrid(CC / 64, ML / 64);
  k_gemm_bf16<<<ggrid, 128, 0, stream>>>(XqB,  WqT, CC, Qh, nullptr, 0);
  k_gemm_bf16<<<ggrid, 128, 0, stream>>>(XkvB, WkT, CC, Kh, nullptr, 1);
  k_gemm_bf16<<<ggrid, 128, 0, stream>>>(XkvB, WvT, CC, Vt, nullptr, 2);

  // Stage 2: logits + pre-softmax mix (WMMA)
  k_logits_premix<<<dim3(LL / 64, LL / 16, BB), 128, 0, stream>>>(Qh, Kh, Wpre, Smix);

  // Stage 3: softmax (in place)
  k_softmax<<<(int)(BB * HH * LL / 8), 256, 0, stream>>>(Smix);

  // Stage 4: post-softmax mix + PV (WMMA)
  k_postmix_pv<<<dim3(LL / 16, BB), 128, 0, stream>>>(Smix, Vt, Wpost, SoB);

  // Stage 5: output projection (WMMA, f32 out)
  k_gemm_bf16<<<ggrid, 128, 0, stream>>>(SoB, WoT, CC, nullptr, (float*)d_out, 3);
}
